// BitNetGQA2_53669911330984
// MI455X (gfx1250) — compile-verified, matrix operand order checked by reading, not run on hardware
//
#include <hip/hip_runtime.h>
#include <hip/hip_bf16.h>
#include <stdint.h>

typedef signed char s8;
typedef __attribute__((ext_vector_type(4)))  int      v4i;
typedef __attribute__((ext_vector_type(8)))  int      v8i;
typedef __attribute__((ext_vector_type(8)))  float    v8f;
typedef __attribute__((ext_vector_type(16))) _Float16 v16h;

#define B_    2
#define S_    2048
#define H_    2048
#define NH_   16
#define NKV_  4
#define HD_   128
#define ROWS_ (B_ * S_)

#if defined(__has_builtin)
#if __has_builtin(__builtin_amdgcn_global_load_async_to_lds_b128) && \
    __has_builtin(__builtin_amdgcn_s_wait_asynccnt)
#define ASYNC_COPY 1
#endif
#endif

#ifdef ASYNC_COPY
typedef __attribute__((address_space(1))) v4i as1_v4i;
typedef __attribute__((address_space(3))) v4i as3_v4i;
__device__ __forceinline__ void async_b128(const void* g, void* l) {
    __builtin_amdgcn_global_load_async_to_lds_b128(
        (as1_v4i*)(uintptr_t)g,
        (as3_v4i*)(uint32_t)(uintptr_t)l,
        0, 0);
}
#endif

__device__ __forceinline__ v8i ld_v8i(const s8* p) {
    v4i lo = *(const v4i*)(p);
    v4i hi = *(const v4i*)(p + 16);
    v8i r;
    r[0] = lo[0]; r[1] = lo[1]; r[2] = lo[2]; r[3] = lo[3];
    r[4] = hi[0]; r[5] = hi[1]; r[6] = hi[2]; r[7] = hi[3];
    return r;
}

__global__ void zero4_kernel(float* p) {
    if (threadIdx.x < 4) p[threadIdx.x] = 0.f;
}

__global__ __launch_bounds__(256) void wabs_kernel(const float* __restrict__ W,
                                                   long n, float* __restrict__ out) {
    __shared__ float red[256];
    const int tid = threadIdx.x;
    float s = 0.f;
    for (long i = (long)blockIdx.x * 256 + tid; i < n; i += (long)gridDim.x * 256)
        s += fabsf(W[i]);
    red[tid] = s;
    __syncthreads();
    for (int off = 128; off > 0; off >>= 1) {
        if (tid < off) red[tid] += red[tid + off];
        __syncthreads();
    }
    if (tid == 0) atomicAdd(out, red[0]);
}

__global__ __launch_bounds__(256) void wquant_kernel(const float* __restrict__ W,
                                                     s8* __restrict__ Wq, long n,
                                                     const float* __restrict__ wsum,
                                                     float numel) {
    const float mean = fmaxf(wsum[0] / numel, 1e-5f);
    const float ws = 1.f / mean;
    for (long i = (long)blockIdx.x * 256 + threadIdx.x; i < n; i += (long)gridDim.x * 256) {
        float v = rintf(W[i] * ws);
        v = fminf(fmaxf(v, -1.f), 1.f);
        Wq[i] = (s8)v;
    }
}

__global__ __launch_bounds__(256) void aquant_kernel(const float* __restrict__ X,
                                                     s8* __restrict__ Xq,
                                                     float* __restrict__ Sout, int K) {
    __shared__ float red[256];
    const int row = blockIdx.x;
    const int tid = threadIdx.x;
    const float* xr = X + (size_t)row * K;
    float mx = 0.f;
    for (int i = tid; i < K; i += 256) mx = fmaxf(mx, fabsf(xr[i]));
    red[tid] = mx;
    __syncthreads();
    for (int off = 128; off > 0; off >>= 1) {
        if (tid < off) red[tid] = fmaxf(red[tid], red[tid + off]);
        __syncthreads();
    }
    const float cmax = fmaxf(red[0], 1e-5f);
    const float s = 127.f / cmax;
    s8* xo = Xq + (size_t)row * K;
    for (int i = tid; i < K; i += 256) {
        float v = rintf(xr[i] * s);
        v = fminf(fmaxf(v, -128.f), 127.f);
        xo[i] = (s8)v;
    }
    if (tid == 0) Sout[row] = cmax / 127.f;
}

__global__ __launch_bounds__(256) void gemm_i8_kernel(
    const s8* __restrict__ A, const s8* __restrict__ Bw,
    const float* __restrict__ xs, const float* __restrict__ wsum, float wnumel,
    float* __restrict__ C, int M, int N, int K) {
    __shared__ __align__(16) s8 As[2][128][64];
    __shared__ __align__(16) s8 Bs[2][64][64];

    const int tid = threadIdx.x;
    const int lane = tid & 31, wave = tid >> 5;
    const int wm = wave >> 1, wn = wave & 1;
    const int m0 = blockIdx.y * 128, n0 = blockIdx.x * 64;
    const int arow = tid >> 1, aseg = tid & 1;
    const int brow = tid >> 2, bseg = tid & 3;
    const int lr = lane & 15, lseg = (lane >> 4) * 32;
    const float meanW = fmaxf(wsum[0] / wnumel, 1e-5f);

    const s8* aB = A + (size_t)(m0 + arow) * K + aseg * 32;
    const s8* bB = Bw + (size_t)(n0 + brow) * K + bseg * 16;

#ifdef ASYNC_COPY
    async_b128(aB,      &As[0][arow][aseg * 32]);
    async_b128(aB + 16, &As[0][arow][aseg * 32 + 16]);
    async_b128(bB,      &Bs[0][brow][bseg * 16]);
    __builtin_amdgcn_s_wait_asynccnt(0);
#else
    {
        v4i a0 = *(const v4i*)(aB);
        v4i a1 = *(const v4i*)(aB + 16);
        v4i b0 = *(const v4i*)(bB);
        *(v4i*)(&As[0][arow][aseg * 32])      = a0;
        *(v4i*)(&As[0][arow][aseg * 32 + 16]) = a1;
        *(v4i*)(&Bs[0][brow][bseg * 16])      = b0;
    }
#endif
    __syncthreads();

    v8i acc[2][2] = {};
    int cur = 0;
    for (int k0 = 0; k0 < K; k0 += 64, cur ^= 1) {
        const int nxt = cur ^ 1;
        const bool more = (k0 + 64) < K;

#ifdef ASYNC_COPY
        if (more) {
            async_b128(aB + k0 + 64, &As[nxt][arow][aseg * 32]);
            async_b128(aB + k0 + 80, &As[nxt][arow][aseg * 32 + 16]);
            async_b128(bB + k0 + 64, &Bs[nxt][brow][bseg * 16]);
        }
#else
        v4i a0, a1, b0;
        if (more) {
            a0 = *(const v4i*)(aB + k0 + 64);
            a1 = *(const v4i*)(aB + k0 + 80);
            b0 = *(const v4i*)(bB + k0 + 64);
        }
#endif
        if (k0 + 128 < K) {
            __builtin_prefetch(aB + k0 + 128, 0, 0);
            __builtin_prefetch(bB + k0 + 128, 0, 0);
        }

        v8i af[2], bf[2];
        for (int i = 0; i < 2; i++) af[i] = ld_v8i(&As[cur][wm * 32 + i * 16 + lr][lseg]);
        for (int j = 0; j < 2; j++) bf[j] = ld_v8i(&Bs[cur][wn * 32 + j * 16 + lr][lseg]);

        for (int i = 0; i < 2; i++)
            for (int j = 0; j < 2; j++)
                acc[i][j] = __builtin_amdgcn_wmma_i32_16x16x64_iu8(
                    true, af[i], true, bf[j], acc[i][j], false, false);

#ifdef ASYNC_COPY
        if (more) __builtin_amdgcn_s_wait_asynccnt(0);
#else
        if (more) {
            *(v4i*)(&As[nxt][arow][aseg * 32])      = a0;
            *(v4i*)(&As[nxt][arow][aseg * 32 + 16]) = a1;
            *(v4i*)(&Bs[nxt][brow][bseg * 16])      = b0;
        }
#endif
        __syncthreads();
    }

    for (int i = 0; i < 2; i++)
        for (int j = 0; j < 2; j++)
            for (int r = 0; r < 8; r++) {
                const int gm = m0 + wm * 32 + i * 16 + r + 8 * (lane >> 4);
                const int gn = n0 + wn * 32 + j * 16 + (lane & 15);
                C[(size_t)gm * N + gn] = (float)acc[i][j][r] * xs[gm] * meanW;
            }
}

__global__ void rope_cast_kernel(const float* __restrict__ X,
                                 _Float16* __restrict__ Y, int nh, long total) {
    long idx = (long)blockIdx.x * blockDim.x + threadIdx.x;
    if (idx >= total) return;
    const int d = (int)(idx & (HD_ - 1));
    const long t = idx / HD_;
    const long row = t / nh;
    const int pos = (int)(row % S_);
    const float inv = __powf(10000.f, -(float)((d & 63) * 2) / 128.f);
    const float ang = (float)pos * inv;
    const float c = __cosf(ang), sn = __sinf(ang);
    const float x = X[idx];
    const float other = (d < 64) ? -X[idx + 64] : X[idx - 64];
    Y[idx] = (_Float16)(x * c + other * sn);
}

__global__ void vtrans_kernel(const float* __restrict__ Vf,
                              _Float16* __restrict__ Vt, long total) {
    long idx = (long)blockIdx.x * blockDim.x + threadIdx.x;
    if (idx >= total) return;
    const int d = (int)(idx & (HD_ - 1));
    const long t = idx / HD_;
    const int kvh = (int)(t % NKV_);
    const long row = t / NKV_;
    const int sp = (int)(row % S_);
    const int b = (int)(row / S_);
    Vt[(((size_t)b * NKV_ + kvh) * HD_ + d) * S_ + sp] = (_Float16)Vf[idx];
}

__global__ __launch_bounds__(128) void attn_kernel(const _Float16* __restrict__ Qh,
                                                   const _Float16* __restrict__ Kh,
                                                   const _Float16* __restrict__ Vt,
                                                   float* __restrict__ O) {
    __shared__ __align__(64) _Float16 Plds[4][16][32];
    const int lane = threadIdx.x & 31, wave = threadIdx.x >> 5;
    const int task = blockIdx.x * 4 + wave;
    const int qb = task & 127;
    const int h  = (task >> 7) & 15;
    const int b  = task >> 11;
    const int kv = h >> 2;
    const int q0 = qb * 16;
    const int lr = lane & 15, lhi = lane >> 4;

    v16h qf[4];
    {
        const size_t qrow = ((size_t)(b * S_ + q0 + lr)) * (NH_ * HD_) + (size_t)h * HD_;
        for (int c = 0; c < 4; c++)
            qf[c] = *(const v16h*)(Qh + qrow + c * 32 + lhi * 16);
    }

    float mi[8], li[8];
    v8f oacc[8] = {};
    for (int r = 0; r < 8; r++) { mi[r] = -1e30f; li[r] = 0.f; }

    const float scale = 0.08838834764831845f;

    for (int k0 = 0; k0 <= q0 + 15; k0 += 32) {
        v8f s0 = {}, s1 = {};
        const size_t kr0 = ((size_t)(b * S_ + k0 + lr)) * (NKV_ * HD_) + (size_t)kv * HD_;
        const size_t kr1 = ((size_t)(b * S_ + k0 + 16 + lr)) * (NKV_ * HD_) + (size_t)kv * HD_;
        for (int c = 0; c < 4; c++) {
            v16h kf0 = *(const v16h*)(Kh + kr0 + c * 32 + lhi * 16);
            v16h kf1 = *(const v16h*)(Kh + kr1 + c * 32 + lhi * 16);
            s0 = __builtin_amdgcn_wmma_f32_16x16x32_f16(false, qf[c], false, kf0,
                                                        (short)0, s0, false, false);
            s1 = __builtin_amdgcn_wmma_f32_16x16x32_f16(false, qf[c], false, kf1,
                                                        (short)0, s1, false, false);
        }

        float sc_old[8];
        for (int r = 0; r < 8; r++) {
            const int mg = q0 + r + 8 * lhi;
            const int n0g = k0 + lr, n1g = n0g + 16;
            float e0 = (n0g <= mg) ? s0[r] * scale : -1e30f;
            float e1 = (n1g <= mg) ? s1[r] * scale : -1e30f;
            float rm = fmaxf(e0, e1);
            rm = fmaxf(rm, __shfl_xor(rm, 1, 32));
            rm = fmaxf(rm, __shfl_xor(rm, 2, 32));
            rm = fmaxf(rm, __shfl_xor(rm, 4, 32));
            rm = fmaxf(rm, __shfl_xor(rm, 8, 32));
            const float mn = fmaxf(mi[r], rm);
            const float so = __expf(mi[r] - mn);
            const float p0 = __expf(e0 - mn);
            const float p1 = __expf(e1 - mn);
            float rs = p0 + p1;
            rs += __shfl_xor(rs, 1, 32);
            rs += __shfl_xor(rs, 2, 32);
            rs += __shfl_xor(rs, 4, 32);
            rs += __shfl_xor(rs, 8, 32);
            li[r] = li[r] * so + rs;
            mi[r] = mn;
            sc_old[r] = so;
            Plds[wave][r + 8 * lhi][lr]      = (_Float16)p0;
            Plds[wave][r + 8 * lhi][16 + lr] = (_Float16)p1;
        }
        for (int j = 0; j < 8; j++)
            for (int r = 0; r < 8; r++) oacc[j][r] *= sc_old[r];

        const v16h pf = *(const v16h*)(&Plds[wave][lr][lhi * 16]);
        const size_t vb = ((size_t)(b * NKV_ + kv) * HD_) * S_;
        for (int j = 0; j < 8; j++) {
            v16h vf = *(const v16h*)(Vt + vb + (size_t)(j * 16 + lr) * S_ + k0 + lhi * 16);
            oacc[j] = __builtin_amdgcn_wmma_f32_16x16x32_f16(false, pf, false, vf,
                                                             (short)0, oacc[j], false, false);
        }
    }

    for (int j = 0; j < 8; j++)
        for (int r = 0; r < 8; r++) {
            const int mg = q0 + r + 8 * lhi;
            O[((size_t)(b * S_ + mg)) * (NH_ * HD_) + (size_t)h * HD_ + j * 16 + lr] =
                oacc[j][r] / li[r];
        }
}

extern "C" void kernel_launch(void* const* d_in, const int* in_sizes, int n_in,
                              void* d_out, int out_size, void* d_ws, size_t ws_size,
                              hipStream_t stream) {
    (void)in_sizes; (void)n_in; (void)out_size; (void)ws_size;
    const float* hidden = (const float*)d_in[0];
    const float* Wq = (const float*)d_in[2];
    const float* Wk = (const float*)d_in[3];
    const float* Wv = (const float*)d_in[4];
    const float* Wo = (const float*)d_in[5];
    float* out = (float*)d_out;

    size_t off = 0;
    auto alloc = [&](size_t bytes) {
        void* p = (char*)d_ws + off;
        off += (bytes + 255) & ~(size_t)255;
        return p;
    };
    float* wsum = (float*)alloc(4 * sizeof(float));
    s8*    xq   = (s8*)alloc((size_t)ROWS_ * H_);
    float* xs   = (float*)alloc((size_t)ROWS_ * sizeof(float));
    s8*    wqQ  = (s8*)alloc((size_t)H_ * H_);
    s8*    wqK  = (s8*)alloc((size_t)(NKV_ * HD_) * H_);
    s8*    wqV  = (s8*)alloc((size_t)(NKV_ * HD_) * H_);
    s8*    wqO  = (s8*)alloc((size_t)H_ * H_);
    float* Qf   = (float*)alloc((size_t)ROWS_ * H_ * sizeof(float));
    float* Kf   = (float*)alloc((size_t)ROWS_ * (NKV_ * HD_) * sizeof(float));
    float* Vf   = (float*)alloc((size_t)ROWS_ * (NKV_ * HD_) * sizeof(float));
    _Float16* Qh = (_Float16*)alloc((size_t)ROWS_ * H_ * sizeof(_Float16));
    _Float16* Kh = (_Float16*)alloc((size_t)ROWS_ * (NKV_ * HD_) * sizeof(_Float16));
    _Float16* Vt = (_Float16*)alloc((size_t)ROWS_ * (NKV_ * HD_) * sizeof(_Float16));
    float* AO = Qf;
    s8*    aoq = xq;
    float* aos = xs;

    const long nQ = (long)H_ * H_;
    const long nK = (long)(NKV_ * HD_) * H_;

    zero4_kernel<<<1, 64, 0, stream>>>(wsum);
    wabs_kernel<<<256, 256, 0, stream>>>(Wq, nQ, wsum + 0);
    wabs_kernel<<<256, 256, 0, stream>>>(Wk, nK, wsum + 1);
    wabs_kernel<<<256, 256, 0, stream>>>(Wv, nK, wsum + 2);
    wabs_kernel<<<256, 256, 0, stream>>>(Wo, nQ, wsum + 3);
    wquant_kernel<<<1024, 256, 0, stream>>>(Wq, wqQ, nQ, wsum + 0, (float)nQ);
    wquant_kernel<<<1024, 256, 0, stream>>>(Wk, wqK, nK, wsum + 1, (float)nK);
    wquant_kernel<<<1024, 256, 0, stream>>>(Wv, wqV, nK, wsum + 2, (float)nK);
    wquant_kernel<<<1024, 256, 0, stream>>>(Wo, wqO, nQ, wsum + 3, (float)nQ);

    aquant_kernel<<<ROWS_, 256, 0, stream>>>(hidden, xq, xs, H_);

    gemm_i8_kernel<<<dim3(H_ / 64, ROWS_ / 128), 256, 0, stream>>>(
        xq, wqQ, xs, wsum + 0, (float)nQ, Qf, ROWS_, H_, H_);
    gemm_i8_kernel<<<dim3((NKV_ * HD_) / 64, ROWS_ / 128), 256, 0, stream>>>(
        xq, wqK, xs, wsum + 1, (float)nK, Kf, ROWS_, NKV_ * HD_, H_);
    gemm_i8_kernel<<<dim3((NKV_ * HD_) / 64, ROWS_ / 128), 256, 0, stream>>>(
        xq, wqV, xs, wsum + 2, (float)nK, Vf, ROWS_, NKV_ * HD_, H_);

    {
        const long tq = (long)ROWS_ * H_;
        const long tk = (long)ROWS_ * (NKV_ * HD_);
        rope_cast_kernel<<<(unsigned)((tq + 255) / 256), 256, 0, stream>>>(Qf, Qh, NH_, tq);
        rope_cast_kernel<<<(unsigned)((tk + 255) / 256), 256, 0, stream>>>(Kf, Kh, NKV_, tk);
        vtrans_kernel<<<(unsigned)((tk + 255) / 256), 256, 0, stream>>>(Vf, Vt, tk);
    }

    attn_kernel<<<1024, 128, 0, stream>>>(Qh, Kh, Vt, AO);

    aquant_kernel<<<ROWS_, 256, 0, stream>>>(AO, aoq, aos, H_);
    gemm_i8_kernel<<<dim3(H_ / 64, ROWS_ / 128), 256, 0, stream>>>(
        aoq, wqO, aos, wsum + 3, (float)nQ, out, ROWS_, H_, H_);
}
